// FaceConvDemodulated_3951369912807
// MI455X (gfx1250) — compile-verified, matrix operand order checked by reading, not run on hardware
//
#include <hip/hip_runtime.h>
#include <hip/hip_bf16.h>

// FaceConvDemodulated for MI455X (gfx1250, wave32, WMMA).
//
// out[n,o] = sum_{k,i} padded[nbr[n,k], i] * (w[o,i,k]*dcoef[o]) + bias[o]
// padded[idx] = (idx == 50000) ? 0 : x[idx]   (faithful to the given setup:
// exactly one pad slot at the end, so rank[i] == i).
//
// Gather-GEMM  C[50000x256] = G[50000x2304] x W[2304x256], as bf16 hi/lo
// split (3x v_wmma_f32_16x16x32_bf16 per tile) for ~fp32 accuracy at
// bf16-matrix-pipe rate. Weights are pre-normalized and pre-swizzled into
// WMMA B-fragment order; the main kernel double-buffers one 32-wide K-chunk
// of B (32 KB) in LDS per block (1 barrier per chunk; next chunk's global
// loads and next A gather overlap the current chunk's WMMAs).
//
// Wave tiling: 2 waves per M-tile, each wave owns 16 rows x 128 outputs
// (8 n-tiles => 64 accumulator VGPRs) to stay within the VGPR budget
// (round 2 showed scratch spills with a 16-n-tile strip).

#define N_FACES 50000
#define C_IN    256
#define C_OUT   256
#define KW      9
#define KTOT    (C_IN * KW)      // 2304 contraction length, flat index k*256+i
#define NCHUNK  (KTOT / 32)      // 72 K-chunks of 32
#define NTILES  ((N_FACES + 15) / 16)  // 3125 M-tiles

typedef __attribute__((ext_vector_type(16))) __bf16        v16bf;
typedef __attribute__((ext_vector_type(8)))  float         v8f;
typedef __attribute__((ext_vector_type(8)))  unsigned int  v8u;

__device__ __forceinline__ unsigned bf16_rne(float v) {
    unsigned u = __builtin_bit_cast(unsigned, v);
    return (u + 0x7FFFu + ((u >> 16) & 1u)) >> 16;   // round-to-nearest-even
}

// ---------------------------------------------------------------------------
// Kernel 1: dcoefs[o] = rsqrt(sum_{i,k} w[o,i,k]^2 + 1e-8)
// ---------------------------------------------------------------------------
__global__ void k_dcoef(const float* __restrict__ w, float* __restrict__ dcoefs) {
    __shared__ float red[256];
    const int o = blockIdx.x;
    float s = 0.f;
    for (int t = threadIdx.x; t < KTOT; t += 256) {
        float v = w[(size_t)o * KTOT + t];
        s += v * v;
    }
    red[threadIdx.x] = s;
    __syncthreads();
    for (int off = 128; off > 0; off >>= 1) {
        if (threadIdx.x < off) red[threadIdx.x] += red[threadIdx.x + off];
        __syncthreads();
    }
    if (threadIdx.x == 0) dcoefs[o] = rsqrtf(red[0] + 1e-8f);
}

// ---------------------------------------------------------------------------
// Kernel 2: normalize weight, bf16 hi/lo split, swizzle into WMMA B-fragment
// order: wsB[((c*16 + j)*32 + lane)*16 + slot], 16-bit operand layout:
//   lanes 0-15 : K offsets {0..7} -> slots 0..7, {16..23} -> slots 8..15
//   lanes 16-31: K offsets {8..15} -> slots 0..7, {24..31} -> slots 8..15
// ---------------------------------------------------------------------------
__global__ void k_pack(const float* __restrict__ w, const float* __restrict__ dcoefs,
                       unsigned short* __restrict__ wh, unsigned short* __restrict__ wl) {
    const int c_flat = blockIdx.x;        // 0..2303  (= k*256 + i)
    const int o      = threadIdx.x;       // 0..255
    const int k = c_flat >> 8;
    const int i = c_flat & 255;

    float val = w[((size_t)o * C_IN + i) * KW + k] * dcoefs[o];
    unsigned hb = bf16_rne(val);
    float    hf = __builtin_bit_cast(float, hb << 16);
    unsigned lb = bf16_rne(val - hf);

    const int c = c_flat >> 5;
    const int r = c_flat & 31;            // K offset within chunk
    const int j = o >> 4;
    const int n = o & 15;
    int laneHalf, slot;
    if      (r < 8)  { laneHalf = 0; slot = r;      }
    else if (r < 16) { laneHalf = 1; slot = r - 8;  }
    else if (r < 24) { laneHalf = 0; slot = r - 8;  }
    else             { laneHalf = 1; slot = r - 16; }
    const int lane = laneHalf * 16 + n;

    const size_t pos = (((size_t)c * 16 + j) * 32 + lane) * 16 + slot;
    wh[pos] = (unsigned short)hb;
    wl[pos] = (unsigned short)lb;
}

// ---------------------------------------------------------------------------
// Staging helpers for the main kernel
// ---------------------------------------------------------------------------
__device__ __forceinline__ void load_b_chunk(const unsigned short* __restrict__ wh,
                                             const unsigned short* __restrict__ wl,
                                             int c, int tid, uint4 rH[4], uint4 rL[4]) {
    const uint4* sH = (const uint4*)(wh + (size_t)c * 8192);
    const uint4* sL = (const uint4*)(wl + (size_t)c * 8192);
#pragma unroll
    for (int q = 0; q < 4; ++q) { rH[q] = sH[tid * 4 + q]; rL[q] = sL[tid * 4 + q]; }
}

__device__ __forceinline__ void store_b_chunk(uint4* dstH, uint4* dstL, int tid,
                                              const uint4 rH[4], const uint4 rL[4]) {
#pragma unroll
    for (int q = 0; q < 4; ++q) { dstH[tid * 4 + q] = rH[q]; dstL[tid * 4 + q] = rL[q]; }
}

__device__ __forceinline__ void load_a_chunk(const float* __restrict__ x,
                                             const long long* __restrict__ nbr,
                                             int row, int c, int b4, float4 g[4]) {
    g[0] = g[1] = g[2] = g[3] = make_float4(0.f, 0.f, 0.f, 0.f);
    if (row < N_FACES) {
        const int k  = c >> 3;
        const int i0 = (c & 7) * 32;
        long long idx = nbr[(long long)row * KW + k];
        if (idx < (long long)N_FACES) {          // idx == 50000 is the zero pad row
            const float4* p = (const float4*)(x + (size_t)idx * C_IN + i0);
            g[0] = p[b4];     g[1] = p[b4 + 1];  // K offsets sel..sel+7
            g[2] = p[b4 + 4]; g[3] = p[b4 + 5];  // K offsets sel+16..sel+23
        }
    }
}

// ---------------------------------------------------------------------------
// Kernel 3: main gather-GEMM. Block = 256 threads = 8 waves = 4 M-tiles x
// 2 N-halves. Wave (mt, nh) owns rows [mtile*16, +16) x cols [nh*128, +128).
// ---------------------------------------------------------------------------
__global__ void __launch_bounds__(256)
k_main(const float* __restrict__ x, const float* __restrict__ bias,
       const long long* __restrict__ nbr,
       const unsigned short* __restrict__ wh, const unsigned short* __restrict__ wl,
       float* __restrict__ out) {
    // double-buffered B chunk: buf b at lds[b*2048], hi at +0, lo at +1024 (64 KB)
    __shared__ uint4 lds[4096];

    const int tid   = threadIdx.x;
    const int lane  = tid & 31;
    const int wave  = tid >> 5;
    const int nh    = wave & 1;            // which 128-wide N half
    const int mtile = blockIdx.x * 4 + (wave >> 1);
    const int n0    = mtile * 16;
    const int m     = lane & 15;
    const int half  = lane >> 4;   // 0: K offsets {0..7,16..23}; 1: {8..15,24..31}
    const int b4    = half * 2;    // float4 offset of this lane's first K run
    const int row   = n0 + m;
    const int j0    = nh * 8;      // first n-tile owned by this wave

    // accumulators: 8 n-tiles x 8 VGPRs, init with bias[o]
    v8f acc[8];
#pragma unroll
    for (int jj = 0; jj < 8; ++jj) {
        float bv = bias[(j0 + jj) * 16 + m];
#pragma unroll
        for (int r = 0; r < 8; ++r) acc[jj][r] = bv;
    }

    // ---- prologue: stage chunk 0 ----
    uint4  rH[4], rL[4];
    float4 ga[4];
    load_b_chunk(wh, wl, 0, tid, rH, rL);
    load_a_chunk(x, nbr, row, 0, b4, ga);
    store_b_chunk(&lds[0], &lds[1024], tid, rH, rL);
    __syncthreads();

    for (int c = 0; c < NCHUNK; ++c) {
        const uint4* curH = &lds[(c & 1) * 2048];
        const uint4* curL = curH + 1024;

        // current chunk's gathered A rows (f32)
        const float4 f0 = ga[0], f1 = ga[1], f2 = ga[2], f3 = ga[3];

        // issue next chunk's global loads early; latency hides under the WMMAs
        const bool more = (c + 1 < NCHUNK);
        if (more) {
            load_b_chunk(wh, wl, c + 1, tid, rH, rL);
            load_a_chunk(x, nbr, row, c + 1, b4, ga);
        }

        // ---- convert current A rows to bf16 hi/lo fragments ----
        float fv[16] = { f0.x, f0.y, f0.z, f0.w, f1.x, f1.y, f1.z, f1.w,
                         f2.x, f2.y, f2.z, f2.w, f3.x, f3.y, f3.z, f3.w };
        unsigned ah[8], al[8];
#pragma unroll
        for (int e = 0; e < 8; ++e) {
            float v0 = fv[2 * e], v1 = fv[2 * e + 1];
            unsigned h0 = bf16_rne(v0), h1 = bf16_rne(v1);
            float r0 = v0 - __builtin_bit_cast(float, h0 << 16);
            float r1 = v1 - __builtin_bit_cast(float, h1 << 16);
            unsigned l0 = bf16_rne(r0), l1 = bf16_rne(r1);
            ah[e] = h0 | (h1 << 16);
            al[e] = l0 | (l1 << 16);
        }
        v16bf Ah = __builtin_bit_cast(v16bf, (v8u){ah[0], ah[1], ah[2], ah[3],
                                                   ah[4], ah[5], ah[6], ah[7]});
        v16bf Al = __builtin_bit_cast(v16bf, (v8u){al[0], al[1], al[2], al[3],
                                                   al[4], al[5], al[6], al[7]});

        // ---- 8 n-tiles, hi/lo split: AhBh + AhBl + AlBh ----
#pragma unroll
        for (int jj = 0; jj < 8; ++jj) {
            const int fi = ((j0 + jj) * 32 + lane) * 2;
            uint4 h0 = curH[fi], h1 = curH[fi + 1];
            uint4 l0 = curL[fi], l1 = curL[fi + 1];
            v16bf Bh = __builtin_bit_cast(v16bf, (v8u){h0.x, h0.y, h0.z, h0.w,
                                                       h1.x, h1.y, h1.z, h1.w});
            v16bf Bl = __builtin_bit_cast(v16bf, (v8u){l0.x, l0.y, l0.z, l0.w,
                                                       l1.x, l1.y, l1.z, l1.w});
            acc[jj] = __builtin_amdgcn_wmma_f32_16x16x32_bf16(false, Ah, false, Bh,
                                                              (short)0, acc[jj], false, false);
            acc[jj] = __builtin_amdgcn_wmma_f32_16x16x32_bf16(false, Ah, false, Bl,
                                                              (short)0, acc[jj], false, false);
            acc[jj] = __builtin_amdgcn_wmma_f32_16x16x32_bf16(false, Al, false, Bh,
                                                              (short)0, acc[jj], false, false);
        }

        // ---- commit next chunk into the other LDS buffer; one barrier/iter ----
        if (more) {
            uint4* nH = &lds[((c + 1) & 1) * 2048];
            store_b_chunk(nH, nH + 1024, tid, rH, rL);
        }
        __syncthreads();
    }

    // ---- store: lane<16 holds col n=lane rows 0..7 (vgpr r),
    //             lane>=16 holds col n=lane-16 rows 8..15 ----
    const int rbase = n0 + half * 8;
#pragma unroll
    for (int jj = 0; jj < 8; ++jj) {
        const int o = (j0 + jj) * 16 + m;
#pragma unroll
        for (int r = 0; r < 8; ++r) {
            const int rr = rbase + r;
            if (rr < N_FACES) out[(size_t)rr * C_OUT + o] = acc[jj][r];
        }
    }
}

// ---------------------------------------------------------------------------
// Host-side launcher
// ---------------------------------------------------------------------------
extern "C" void kernel_launch(void* const* d_in, const int* in_sizes, int n_in,
                              void* d_out, int out_size, void* d_ws, size_t ws_size,
                              hipStream_t stream) {
    const float*     x    = (const float*)d_in[0];      // [50000,256]
    const float*     w    = (const float*)d_in[1];      // [256,256,1,9]
    const float*     bias = (const float*)d_in[2];      // [256]
    const long long* nbr  = (const long long*)d_in[3];  // [50000,9] int64
    // d_in[4] face_is_pad / d_in[5] pad_size: only last slot is pad -> folded
    // into the idx == 50000 check in k_main.

    // workspace layout: [0,1KB) dcoefs | B-hi bf16 (1.125 MB) | B-lo bf16
    float*          dcoefs = (float*)d_ws;
    unsigned short* wh     = (unsigned short*)((char*)d_ws + 1024);
    unsigned short* wl     = wh + (size_t)KTOT * C_OUT;

    k_dcoef<<<dim3(C_OUT), dim3(256), 0, stream>>>(w, dcoefs);
    k_pack <<<dim3(KTOT),  dim3(256), 0, stream>>>(w, dcoefs, wh, wl);

    const int nblocks = (NTILES + 3) / 4;               // 782
    k_main <<<dim3(nblocks), dim3(256), 0, stream>>>(x, bias, nbr, wh, wl,
                                                     (float*)d_out);
    (void)in_sizes; (void)n_in; (void)out_size; (void)ws_size;
}
